// Generator_30666066493831
// MI455X (gfx1250) — matrix operand order repaired and checked by reading, not compiled, not run on hardware
//
#include <hip/hip_runtime.h>

typedef __attribute__((ext_vector_type(16))) _Float16 v16h;
typedef __attribute__((ext_vector_type(8)))  float    v8f;

#define SEQ  20
#define FEAT 5
#define NU1  10
#define NU2  7
#define NU3  4
#define ODIM 4
#define NTILES 13                     // 4+4+4 gate tiles + 1 dense tile
#define WAVES_PER_BLOCK 4
#define ROWS_PER_WAVE   16
#define ROWS_PER_BLOCK  (WAVES_PER_BLOCK * ROWS_PER_WAVE)   // 64
#define BLOCK_THREADS   (WAVES_PER_BLOCK * 32)              // 128

// ---- activations on the TRANS pipe ----
__device__ __forceinline__ float fast_tanh(float x) {
#if __has_builtin(__builtin_amdgcn_tanhf)
  return __builtin_amdgcn_tanhf(x);                 // v_tanh_f32 (1 TRANS)
#else
  float e = __builtin_amdgcn_exp2f(2.8853900817779268f * x);
  return 1.0f - 2.0f * __builtin_amdgcn_rcpf(e + 1.0f);
#endif
}
__device__ __forceinline__ float fast_sig(float x) {
#if __has_builtin(__builtin_amdgcn_tanhf)
  return __builtin_fmaf(0.5f, __builtin_amdgcn_tanhf(0.5f * x), 0.5f);
#else
  float e = __builtin_amdgcn_exp2f(-1.4426950408889634f * x);
  return __builtin_amdgcn_rcpf(1.0f + e);
#endif
}

__device__ __forceinline__ v8f splat8(float v) {
  v8f r;
#pragma unroll
  for (int e = 0; e < 8; ++e) r[e] = v;
  return r;
}

__device__ __forceinline__ v8f wmma_f16(v16h a, v16h b, v8f c) {
  return __builtin_amdgcn_wmma_f32_16x16x32_f16(false, a, false, b, (short)0, c, false, false);
}

// A fragment from a [16 rows][32 K] f16 LDS buffer, per CDNA5 16-bit A layout:
// lanes 0-15 M=lane (K 0-7 in v0-3, 16-23 in v4-7); lanes 16-31 K 8-15 / 24-31.
__device__ __forceinline__ v16h load_afrag(const _Float16* buf, int m, int half) {
  union { v16h v; unsigned u[8]; } r;
  const unsigned* p = (const unsigned*)(buf + m * 32);
#pragma unroll
  for (int j = 0; j < 4; ++j) r.u[j] = p[4 * half + j];
#pragma unroll
  for (int j = 0; j < 4; ++j) r.u[4 + j] = p[8 + 4 * half + j];
  return r.v;
}

// B fragment from LDS tile stored [col][k] (col-major-by-lane, K contiguous):
// lane n = column, K 8*half..+7 in v0-3, 16+8*half..+7 in v4-7 (matches A).
__device__ __forceinline__ v16h load_bfrag(const _Float16* tile, int n, int half) {
  union { v16h v; unsigned u[8]; } r;
  const unsigned* p = (const unsigned*)(tile + n * 32);
#pragma unroll
  for (int j = 0; j < 4; ++j) r.u[j] = p[4 * half + j];
#pragma unroll
  for (int j = 0; j < 4; ++j) r.u[4 + j] = p[8 + 4 * half + j];
  return r.v;
}

// One LSTM layer step for 16 batch rows: 4 gate-padded WMMA tiles then
// per-lane gate math (lane = unit, accumulator element v = batch row).
__device__ __forceinline__ void lstm_layer_step(
    v16h a, const _Float16* Bt, const float* bias, float* cs,
    int n, int half, int U,
    _Float16* d0, int o0, _Float16* d1, int o1)
{
  v8f zi = wmma_f16(a, load_bfrag(Bt + 0 * 512, n, half), splat8(bias[0]));
  v8f zf = wmma_f16(a, load_bfrag(Bt + 1 * 512, n, half), splat8(bias[1]));
  v8f zg = wmma_f16(a, load_bfrag(Bt + 2 * 512, n, half), splat8(bias[2]));
  v8f zo = wmma_f16(a, load_bfrag(Bt + 3 * 512, n, half), splat8(bias[3]));
#pragma unroll
  for (int v = 0; v < 8; ++v) {
    float ig = fast_sig(zi[v]);
    float fg = fast_sig(zf[v]);
    float gg = fast_tanh(zg[v]);
    float og = fast_sig(zo[v]);
    float c  = fg * cs[v] + ig * gg;
    cs[v] = c;
    float h = og * fast_tanh(c);
    if (n < U) {
      int m = v + 8 * half;
      d0[m * 32 + o0 + n] = (_Float16)h;
      if (d1) d1[m * 32 + o1 + n] = (_Float16)h;
    }
  }
}

__global__ __launch_bounds__(BLOCK_THREADS)
void lstm_stack_kernel(
    const float* __restrict__ x,
    const float* __restrict__ Wk1, const float* __restrict__ Wr1, const float* __restrict__ b1,
    const float* __restrict__ Wk2, const float* __restrict__ Wr2, const float* __restrict__ b2,
    const float* __restrict__ Wk3, const float* __restrict__ Wr3, const float* __restrict__ b3,
    const float* __restrict__ Wd,  const float* __restrict__ bd,
    float* __restrict__ out, int batch)
{
  __shared__ _Float16 xs[ROWS_PER_BLOCK * SEQ * FEAT];      // 12.8 KB
  __shared__ _Float16 abuf[WAVES_PER_BLOCK][3][512];        // 12.3 KB
  __shared__ _Float16 bfrag[NTILES * 512];                  // 13.3 KB

  const int tid  = threadIdx.x;
  const int wave = tid >> 5;
  const int lane = tid & 31;
  const int n    = lane & 15;   // unit / output column
  const int half = lane >> 4;
  const int blockRow = blockIdx.x * ROWS_PER_BLOCK;
  const int rowsHere = min(ROWS_PER_BLOCK, batch - blockRow);

  // ---- stage x for this block (64 contiguous rows x 100 f32), coalesced ----
  {
    const float* xg = x + (size_t)blockRow * (SEQ * FEAT);
    for (int i = tid; i < rowsHere * SEQ * FEAT; i += BLOCK_THREADS)
      xs[i] = (_Float16)xg[i];
  }
  // ---- zero A staging buffers (pad regions must stay exactly zero) ----
  {
    _Float16* f = &abuf[0][0][0];
    for (int i = tid; i < WAVES_PER_BLOCK * 3 * 512; i += BLOCK_THREADS)
      f[i] = (_Float16)0.0f;
  }
  // ---- build B tiles in LDS, fragment-friendly layout [tile][col][k] ----
  // tiles 0-3: layer1 gates i,f,g,o (K rows: Wk1(0-4), Wr1(5-14))
  // tiles 4-7: layer2           (K rows: Wk2(0-9), Wr2(10-16))
  // tiles 8-11: layer3          (K rows: Wk3(0-6), Wr3(7-10))
  // tile 12: dense Wd at K rows 7-10 (shares layer-3 staging buffer)
  for (int i = tid; i < NTILES * 512; i += BLOCK_THREADS) {
    int tile = i >> 9, rem = i & 511, col = rem >> 5, k = rem & 31;
    float w = 0.0f;
    if (tile < 4) {
      if (col < NU1) {
        int c = tile * NU1 + col;
        if (k < FEAT)            w = Wk1[k * 4 * NU1 + c];
        else if (k < FEAT + NU1) w = Wr1[(k - FEAT) * 4 * NU1 + c];
      }
    } else if (tile < 8) {
      if (col < NU2) {
        int c = (tile - 4) * NU2 + col;
        if (k < NU1)             w = Wk2[k * 4 * NU2 + c];
        else if (k < NU1 + NU2)  w = Wr2[(k - NU1) * 4 * NU2 + c];
      }
    } else if (tile < 12) {
      if (col < NU3) {
        int c = (tile - 8) * NU3 + col;
        if (k < NU2)             w = Wk3[k * 4 * NU3 + c];
        else if (k < NU2 + NU3)  w = Wr3[(k - NU2) * 4 * NU3 + c];
      }
    } else {
      if (col < ODIM && k >= NU2 && k < NU2 + NU3)
        w = Wd[(k - NU2) * ODIM + col];
    }
    bfrag[(tile << 9) + (col << 5) + k] = (_Float16)w;
  }

  // ---- per-lane biases (ride in the WMMA C operand) ----
  float bi1[4], bi2[4], bi3[4], bid;
#pragma unroll
  for (int g = 0; g < 4; ++g) {
    bi1[g] = (n < NU1) ? b1[g * NU1 + n] : 0.0f;
    bi2[g] = (n < NU2) ? b2[g * NU2 + n] : 0.0f;
    bi3[g] = (n < NU3) ? b3[g * NU3 + n] : 0.0f;
  }
  bid = (n < ODIM) ? bd[n] : 0.0f;

  // cell states: lane = unit, element v = batch row (v + 8*half)
  float c1s[8] = {0}, c2s[8] = {0}, c3s[8] = {0};

  _Float16* A1 = abuf[wave][0];   // [x_t(0..4) | h1(5..14)]
  _Float16* A2 = abuf[wave][1];   // [h1(0..9)  | h2(10..16)]
  _Float16* A3 = abuf[wave][2];   // [h2(0..6)  | h3(7..10)] (also dense input)
  __syncthreads();

  for (int t = 0; t < SEQ; ++t) {
    // stage x_t into A1 cols [0,5)
    if (lane < 16) {
      int r = wave * 16 + lane;
#pragma unroll
      for (int f = 0; f < FEAT; ++f)
        A1[lane * 32 + f] = xs[(r * SEQ + t) * FEAT + f];
    }
    __syncthreads();

    v16h a = load_afrag(A1, n, half);
    lstm_layer_step(a, bfrag + 0 * 512, bi1, c1s, n, half, NU1, A1, FEAT, A2, 0);
    __syncthreads();

    a = load_afrag(A2, n, half);
    lstm_layer_step(a, bfrag + 4 * 512, bi2, c2s, n, half, NU2, A2, NU1, A3, 0);
    __syncthreads();

    a = load_afrag(A3, n, half);
    lstm_layer_step(a, bfrag + 8 * 512, bi3, c3s, n, half, NU3, A3, NU2, nullptr, 0);
    __syncthreads();

    // dense head: A3 = [h2(t) | h3(t)], Wd lives at K rows 7..10 of tile 12
    a = load_afrag(A3, n, half);
    v8f zd = wmma_f16(a, load_bfrag(bfrag + 12 * 512, n, half), splat8(bid));
    if (n < ODIM) {
#pragma unroll
      for (int v = 0; v < 8; ++v) {
        int m = v + 8 * half;
        int row = blockRow + wave * 16 + m;
        if (row < batch)
          out[(size_t)(row * SEQ + t) * ODIM + n] = fast_sig(zd[v]);
      }
    }
    __syncthreads();
  }
}

extern "C" void kernel_launch(void* const* d_in, const int* in_sizes, int n_in,
                              void* d_out, int out_size, void* d_ws, size_t ws_size,
                              hipStream_t stream) {
  (void)n_in; (void)out_size; (void)d_ws; (void)ws_size;
  const float* x   = (const float*)d_in[0];
  const float* Wk1 = (const float*)d_in[1];
  const float* Wr1 = (const float*)d_in[2];
  const float* b1  = (const float*)d_in[3];
  const float* Wk2 = (const float*)d_in[4];
  const float* Wr2 = (const float*)d_in[5];
  const float* b2  = (const float*)d_in[6];
  const float* Wk3 = (const float*)d_in[7];
  const float* Wr3 = (const float*)d_in[8];
  const float* b3  = (const float*)d_in[9];
  const float* Wd  = (const float*)d_in[10];
  const float* bd  = (const float*)d_in[11];

  int batch = in_sizes[0] / (SEQ * FEAT);
  int grid  = (batch + ROWS_PER_BLOCK - 1) / ROWS_PER_BLOCK;

  lstm_stack_kernel<<<grid, BLOCK_THREADS, 0, stream>>>(
      x, Wk1, Wr1, b1, Wk2, Wr2, b2, Wk3, Wr3, b3, Wd, bd,
      (float*)d_out, batch);
}
